// BaselineAttention_52707838657216
// MI455X (gfx1250) — matrix operand order repaired and checked by reading, not compiled, not run on hardware
//
#include <hip/hip_runtime.h>
#include <hip/hip_bf16.h>
#include <stdint.h>

// ---------------------------------------------------------------------------
// Causal MHA forward for MI455X (gfx1250).
// GEMMs: LDS-staged, double-buffered block tiles feeding v_wmma_f32_16x16x32_bf16.
// Attention: register-pipelined flash attention on WMMA.
// B=2, S=2048, D=1024, H=16, HD=64.
// ---------------------------------------------------------------------------

#define Bb 2
#define Ss 2048
#define Dd 1024
#define Hh 16
#define HD 64
#define LDB 40  // LDS row stride (elements) for the 64x32 B tile, pads banks

typedef __attribute__((ext_vector_type(16))) __bf16 v16bf;
typedef __attribute__((ext_vector_type(8)))  float  v8f;

union V16U { uint4 u[2]; v16bf h; };

__device__ __forceinline__ v8f wmma_bf(v16bf a, v16bf b, v8f c) {
  // D = A(16x32) * B(32x16) + C, f32 accumulate
  return __builtin_amdgcn_wmma_f32_16x16x32_bf16(false, a, false, b, (short)0, c,
                                                 false, false);
}

// A-fragment (16x32, bf16, row-major source, leading dim ld):
// lane m (0-15) holds row m; elements 0-15 -> K = k0 + half*16 .. +15
// (half = lane>>4)
__device__ __forceinline__ v16bf load_a_g(const __bf16* __restrict__ base, int ld,
                                          int row0, int k0, int lane) {
  const __bf16* p = base + (size_t)(row0 + (lane & 15)) * ld + k0 + ((lane >> 4) << 4);
  V16U t;
  t.u[0] = *(const uint4*)p;
  t.u[1] = *(const uint4*)(p + 8);
  return t.h;
}

// B-fragment (32x16, bf16) from global where column n of B is a contiguous row
// of `base` (K rows / V^T rows): lane n (0-15) holds K=k0..k0+15, lane n+16
// holds K=k0+16..k0+31 -> one 32-byte contiguous load per lane.
__device__ __forceinline__ v16bf load_b_t(const __bf16* __restrict__ base, int ld,
                                          int col0, int k0, int lane) {
  const __bf16* p = base + (size_t)(col0 + (lane & 15)) * ld + k0 + ((lane >> 4) << 4);
  V16U t;
  t.u[0] = *(const uint4*)p;
  t.u[1] = *(const uint4*)(p + 8);
  return t.h;
}

// B-fragment from an LDS-staged tile: rows = B columns, row stride LDB elems
__device__ __forceinline__ v16bf load_b_lds(const __bf16* bs, int col0, int lane) {
  const __bf16* p = bs + (size_t)(col0 + (lane & 15)) * LDB + ((lane >> 4) << 4);
  V16U t;
  t.u[0] = *(const uint4*)p;
  t.u[1] = *(const uint4*)(p + 8);
  return t.h;
}

// A-fragment from an LDS tile of 16 rows x 32 cols (row stride 32 elems)
__device__ __forceinline__ v16bf load_a_lds(const __bf16* lds, int lane) {
  const __bf16* p = lds + (lane & 15) * 32 + ((lane >> 4) << 4);
  V16U t;
  t.u[0] = *(const uint4*)p;
  t.u[1] = *(const uint4*)(p + 8);
  return t.h;
}

__device__ __forceinline__ float red_max16(float v) {
#pragma unroll
  for (int m = 1; m <= 8; m <<= 1) v = fmaxf(v, __shfl_xor(v, m, 32));
  return v;
}
__device__ __forceinline__ float red_sum16(float v) {
#pragma unroll
  for (int m = 1; m <= 8; m <<= 1) v += __shfl_xor(v, m, 32);
  return v;
}

// ---------------------------------------------------------------------------
// f32 -> bf16 conversion, 4 elements/thread
// ---------------------------------------------------------------------------
__global__ void cvt_f32_bf16(const float* __restrict__ in, __bf16* __restrict__ out,
                             int n4) {
  int i = blockIdx.x * blockDim.x + threadIdx.x;
  if (i < n4) {
    float4 v = ((const float4*)in)[i];
    out[i * 4 + 0] = (__bf16)v.x;
    out[i * 4 + 1] = (__bf16)v.y;
    out[i * 4 + 2] = (__bf16)v.z;
    out[i * 4 + 3] = (__bf16)v.w;
  }
}

// ---------------------------------------------------------------------------
// LDS-staged, double-buffered 128x64 block GEMM core (256 threads = 8 waves).
// B tile (64 cols x 32 k = 4KB) loaded cooperatively ONCE per block per k-step
// (8x traffic cut vs per-wave loads), staged through 2 LDS buffers so the next
// stage's global load is in flight while WMMAs consume the current stage from
// LDS. A fragments are per-wave register double-buffered from global.
// acc[] is this wave's 16x64 accumulator tile.
// ---------------------------------------------------------------------------
__device__ __forceinline__ void gemm_core_lds(const __bf16* __restrict__ A,
                                              const __bf16* __restrict__ W,
                                              int mrow_blk, int ncol, int tid,
                                              __bf16* ldsB, v8f acc[4]) {
  const int lane = tid & 31;
  const int wv = tid >> 5;
  const int mrow = mrow_blk + wv * 16;

  // cooperative B loader: thread covers (row = tid>>2, 16B chunk = tid&3)
  const int brow = tid >> 2;         // 0..63 (B column / W row)
  const int bchk = (tid & 3) << 3;   // element offset 0,8,16,24 within k-step
  const __bf16* wptr = W + (size_t)(ncol + brow) * Dd + bchk;
  __bf16* lrow = ldsB + (size_t)brow * LDB + bchk;
  const int stageSz = 64 * LDB;

  // ---- prologue: stage 0 ----
  uint4 breg = *(const uint4*)(wptr);
  v16bf a_c = load_a_g(A, Dd, mrow, 0, lane);
  *(uint4*)(lrow) = breg;
  __syncthreads();

  int st = 0;
  for (int k0 = 0; k0 < Dd; k0 += 32, st ^= 1) {
    const int kn = (k0 + 32 < Dd) ? k0 + 32 : k0;  // branch-free pipeline tail

    // prefetch ~8 k-steps ahead into L2 (global_prefetch_b8)
    if (k0 + 256 < Dd) __builtin_prefetch(wptr + k0 + 256, 0, 1);

    // issue next-stage global loads first (in flight during compute)
    uint4 bnext = *(const uint4*)(wptr + kn);
    v16bf a_n = load_a_g(A, Dd, mrow, kn, lane);

    // fragments from the resident LDS stage
    const __bf16* bs = ldsB + st * stageSz;
    v16bf b0 = load_b_lds(bs, 0, lane);
    v16bf b1 = load_b_lds(bs, 16, lane);
    v16bf b2 = load_b_lds(bs, 32, lane);
    v16bf b3 = load_b_lds(bs, 48, lane);

    acc[0] = wmma_bf(a_c, b0, acc[0]);
    acc[1] = wmma_bf(a_c, b1, acc[1]);
    acc[2] = wmma_bf(a_c, b2, acc[2]);
    acc[3] = wmma_bf(a_c, b3, acc[3]);

    __syncthreads();                        // all waves done reading old stage
    *(uint4*)(lrow + (st ^ 1) * stageSz) = bnext;
    __syncthreads();                        // stage (st^1) visible to all
    a_c = a_n;
  }
}

// ---------------------------------------------------------------------------
// QKV projection: qkv[M=4096, N=3072] = X[M,1024] * Wqkv[N,1024]^T + b
// Epilogue splits into Q (scaled by 1/8) [B,H,S,64], K [B,H,S,64], V^T [B,H,64,S]
// ---------------------------------------------------------------------------
__global__ void gemm_qkv(const __bf16* __restrict__ X, const __bf16* __restrict__ W,
                         const float* __restrict__ bias, __bf16* __restrict__ Q,
                         __bf16* __restrict__ K, __bf16* __restrict__ Vt) {
  __shared__ __attribute__((aligned(16))) __bf16 ldsB[2][64][LDB];
  const int lane = threadIdx.x & 31;
  const int wv = threadIdx.x >> 5;
  const int mrow = blockIdx.y * 128 + wv * 16;
  const int ncol = blockIdx.x * 64;

  v8f z = {};
  v8f acc[4];
#pragma unroll
  for (int t = 0; t < 4; ++t) acc[t] = z;

  gemm_core_lds(X, W, blockIdx.y * 128, ncol, threadIdx.x, &ldsB[0][0][0], acc);

#pragma unroll
  for (int t = 0; t < 4; ++t) {
    const int n = ncol + t * 16 + (lane & 15);
    const float bv = bias[n];
#pragma unroll
    for (int r = 0; r < 8; ++r) {
      const int m = mrow + ((lane >> 4) << 3) + r;
      const int s = m & (Ss - 1);
      const int bz = m >> 11;
      const float val = acc[t][r] + bv;
      if (n < Dd) {  // Q, fold in 1/sqrt(HD)=0.125
        const int hh = n >> 6, d = n & 63;
        Q[(((size_t)(bz * Hh + hh)) * Ss + s) * HD + d] = (__bf16)(val * 0.125f);
      } else if (n < 2 * Dd) {  // K
        const int nn = n - Dd, hh = nn >> 6, d = nn & 63;
        K[(((size_t)(bz * Hh + hh)) * Ss + s) * HD + d] = (__bf16)val;
      } else {  // V transposed: [B,H,64,S]
        const int nn = n - 2 * Dd, hh = nn >> 6, d = nn & 63;
        Vt[(((size_t)(bz * Hh + hh)) * HD + d) * Ss + s] = (__bf16)val;
      }
    }
  }
}

// ---------------------------------------------------------------------------
// Flash attention: 4 waves/block, each wave owns a 16-row query tile.
// Pipelined: V frags issued before score WMMAs; next K frags issued before
// softmax so they overlap VALU + PV WMMAs.
// ---------------------------------------------------------------------------
__global__ void attn(const __bf16* __restrict__ Q, const __bf16* __restrict__ K,
                     const __bf16* __restrict__ Vt,
                     const unsigned char* __restrict__ amask,
                     __bf16* __restrict__ Oa) {
  const int lane = threadIdx.x & 31;
  const int wv = threadIdx.x >> 5;
  const int bz = blockIdx.z;
  const int h = blockIdx.y;
  const int qbase = blockIdx.x * 64 + wv * 16;

  const __bf16* Qh = Q + ((size_t)(bz * Hh + h)) * Ss * HD;
  const __bf16* Kh = K + ((size_t)(bz * Hh + h)) * Ss * HD;
  const __bf16* Vth = Vt + ((size_t)(bz * Hh + h)) * HD * Ss;
  const unsigned char* mk = amask + (size_t)bz * Ss;

  __shared__ __attribute__((aligned(16))) __bf16 lds[4][16][32];

  // Q tile fragments (16 rows x 64 head-dim = two K=32 fragments), loop-invariant
  const v16bf aq0 = load_a_g(Qh, HD, qbase, 0, lane);
  const v16bf aq1 = load_a_g(Qh, HD, qbase, 32, lane);

  const float NEG = -3.0e38f;
  float m_i[8], l_i[8];
#pragma unroll
  for (int r = 0; r < 8; ++r) { m_i[r] = NEG; l_i[r] = 0.f; }
  v8f z = {};
  v8f o[4];
#pragma unroll
  for (int t = 0; t < 4; ++t) o[t] = z;

  const int qmax = qbase + 15;
  const int kend = ((qmax + 32) >> 5) << 5;  // ceil((qmax+1)/32)*32
  const int nl = lane & 15;
  const int mloc = (lane >> 4) << 3;  // local row base for this half-wave

  // preload K fragments for block 0
  v16bf bk0 = load_b_t(Kh, HD, 0, 0, lane);
  v16bf bk1 = load_b_t(Kh, HD, 0, 32, lane);
  v16bf bk2 = load_b_t(Kh, HD, 16, 0, lane);
  v16bf bk3 = load_b_t(Kh, HD, 16, 32, lane);

  for (int kb32 = 0; kb32 < kend; kb32 += 32) {
    // issue V fragments for this block early (consumed only after softmax)
    v16bf bv0 = load_b_t(Vth, Ss, 0, kb32, lane);
    v16bf bv1 = load_b_t(Vth, Ss, 16, kb32, lane);
    v16bf bv2 = load_b_t(Vth, Ss, 32, kb32, lane);
    v16bf bv3 = load_b_t(Vth, Ss, 48, kb32, lane);

    // issue next block's K fragments (branch-free clamp on the tail)
    const int kb_n = (kb32 + 32 < kend) ? kb32 + 32 : kb32;
    v16bf nk0 = load_b_t(Kh, HD, kb_n, 0, lane);
    v16bf nk1 = load_b_t(Kh, HD, kb_n, 32, lane);
    v16bf nk2 = load_b_t(Kh, HD, kb_n + 16, 0, lane);
    v16bf nk3 = load_b_t(Kh, HD, kb_n + 16, 32, lane);

    // ---- scores: 16x32 tile of Q*K^T (scale pre-folded into Q) ----
    v8f s0 = z, s1 = z;
    s0 = wmma_bf(aq0, bk0, s0);
    s0 = wmma_bf(aq1, bk1, s0);
    s1 = wmma_bf(aq0, bk2, s1);
    s1 = wmma_bf(aq1, bk3, s1);

    const int key0 = kb32 + nl;
    const int key1 = kb32 + 16 + nl;
    const bool km0 = mk[key0] != 0;
    const bool km1 = mk[key1] != 0;

    float scale_f[8];
#pragma unroll
    for (int r = 0; r < 8; ++r) {
      const int row = qbase + mloc + r;
      float v0 = s0[r], v1 = s1[r];
      if (key0 > row || !km0) v0 = NEG;
      if (key1 > row || !km1) v1 = NEG;
      const float mb = red_max16(fmaxf(v0, v1));
      const float mn = fmaxf(m_i[r], mb);
      const float sc = __expf(m_i[r] - mn);
      const float p0 = __expf(v0 - mn);
      const float p1 = __expf(v1 - mn);
      lds[wv][mloc + r][nl] = (__bf16)p0;
      lds[wv][mloc + r][16 + nl] = (__bf16)p1;
      l_i[r] = l_i[r] * sc + red_sum16(p0 + p1);
      m_i[r] = mn;
      scale_f[r] = sc;
    }

    // rescale running output by exp(m_old - m_new)
#pragma unroll
    for (int t = 0; t < 4; ++t)
#pragma unroll
      for (int r = 0; r < 8; ++r) o[t][r] *= scale_f[r];

    // ---- PV: P(16x32) * V(32x64) ----
    const v16bf pa = load_a_lds(&lds[wv][0][0], lane);
    o[0] = wmma_bf(pa, bv0, o[0]);
    o[1] = wmma_bf(pa, bv1, o[1]);
    o[2] = wmma_bf(pa, bv2, o[2]);
    o[3] = wmma_bf(pa, bv3, o[3]);

    bk0 = nk0; bk1 = nk1; bk2 = nk2; bk3 = nk3;
  }

  // normalize + scatter to [B,S,D] bf16 (per-head slice) for the out-proj GEMM
#pragma unroll
  for (int t = 0; t < 4; ++t)
#pragma unroll
    for (int r = 0; r < 8; ++r) {
      const int row = qbase + mloc + r;
      const int d = h * HD + t * 16 + nl;
      Oa[((size_t)(bz * Ss + row)) * Dd + d] = (__bf16)(o[t][r] / l_i[r]);
    }
}

// ---------------------------------------------------------------------------
// Output projection: out[M=4096, N=1024] = A[M,1024] * Wout[N,1024]^T + b (f32)
// ---------------------------------------------------------------------------
__global__ void gemm_out(const __bf16* __restrict__ A, const __bf16* __restrict__ W,
                         const float* __restrict__ bias, float* __restrict__ out) {
  __shared__ __attribute__((aligned(16))) __bf16 ldsB[2][64][LDB];
  const int lane = threadIdx.x & 31;
  const int wv = threadIdx.x >> 5;
  const int mrow = blockIdx.y * 128 + wv * 16;
  const int ncol = blockIdx.x * 64;

  v8f z = {};
  v8f acc[4];
#pragma unroll
  for (int t = 0; t < 4; ++t) acc[t] = z;

  gemm_core_lds(A, W, blockIdx.y * 128, ncol, threadIdx.x, &ldsB[0][0][0], acc);

#pragma unroll
  for (int t = 0; t < 4; ++t) {
    const int n = ncol + t * 16 + (lane & 15);
    const float bv = bias[n];
#pragma unroll
    for (int r = 0; r < 8; ++r) {
      const int m = mrow + ((lane >> 4) << 3) + r;
      out[(size_t)m * Dd + n] = acc[t][r] + bv;
    }
  }
}

// ---------------------------------------------------------------------------
extern "C" void kernel_launch(void* const* d_in, const int* in_sizes, int n_in,
                              void* d_out, int out_size, void* d_ws, size_t ws_size,
                              hipStream_t stream) {
  const float* x = (const float*)d_in[0];
  const unsigned char* amask = (const unsigned char*)d_in[1];
  const float* Wqkv = (const float*)d_in[2];
  const float* bqkv = (const float*)d_in[3];
  const float* Wout = (const float*)d_in[4];
  const float* bout = (const float*)d_in[5];
  float* out = (float*)d_out;

  char* ws = (char*)d_ws;
  const size_t nX = (size_t)Bb * Ss * Dd;        // 4,194,304
  const size_t nWq = (size_t)3 * Dd * Dd;        // 3,145,728
  const size_t nWo = (size_t)Dd * Dd;            // 1,048,576
  const size_t nQ = (size_t)Bb * Hh * Ss * HD;   // 4,194,304

  __bf16* xb = (__bf16*)(ws);
  __bf16* wqb = (__bf16*)(ws + 2 * nX);
  __bf16* wob = (__bf16*)(ws + 2 * (nX + nWq));
  __bf16* Qp = (__bf16*)(ws + 2 * (nX + nWq + nWo));
  __bf16* Kp = (__bf16*)(ws + 2 * (nX + nWq + nWo + nQ));
  __bf16* Vtp = (__bf16*)(ws + 2 * (nX + nWq + nWo + 2 * nQ));
  __bf16* Ab = (__bf16*)(ws + 2 * (nX + nWq + nWo + 3 * nQ));

  // 1) downconvert inputs to bf16
  cvt_f32_bf16<<<(int)((nX / 4 + 255) / 256), 256, 0, stream>>>(x, xb, (int)(nX / 4));
  cvt_f32_bf16<<<(int)((nWq / 4 + 255) / 256), 256, 0, stream>>>(Wqkv, wqb, (int)(nWq / 4));
  cvt_f32_bf16<<<(int)((nWo / 4 + 255) / 256), 256, 0, stream>>>(Wout, wob, (int)(nWo / 4));

  // 2) QKV projection (M=4096, N=3072, K=1024)
  gemm_qkv<<<dim3(3 * Dd / 64, Bb * Ss / 128), 256, 0, stream>>>(xb, wqb, bqkv, Qp, Kp, Vtp);

  // 3) flash attention: grid (S/64, H, B), 4 waves/block
  attn<<<dim3(Ss / 64, Hh, Bb), 128, 0, stream>>>(Qp, Kp, Vtp, amask, Ab);

  // 4) output projection (M=4096, N=1024, K=1024) -> f32 d_out
  gemm_out<<<dim3(Dd / 64, Bb * Ss / 128), 256, 0, stream>>>(Ab, wob, bout, out);
}
